// TTMS_75935021793625
// MI455X (gfx1250) — matrix operand order repaired and checked by reading, not compiled, not run on hardware
//
#include <hip/hip_runtime.h>
#include <hip/hip_bf16.h>

// ---------------------------------------------------------------------------
// Problem constants (from the reference)
// ---------------------------------------------------------------------------
#define BB   8
#define CDD  5
#define HIS  50
#define SS   64
#define DD   768
#define NHH  12
#define DH   64
#define FFF  3072
#define TOPK 8
#define US   (HIS * TOPK)   // 400, user-BERT sequence length
#define RT_D_INV 0.036084391824351613f   // 1/sqrt(768)

typedef __attribute__((ext_vector_type(16))) __bf16 v16bf;
typedef __attribute__((ext_vector_type(8)))  __bf16 v8bf;
typedef __attribute__((ext_vector_type(8)))  float  v8f;

__device__ __forceinline__ __bf16 f2bf(float f) {
  unsigned u = __builtin_bit_cast(unsigned, f);
  u += 0x7FFFu + ((u >> 16) & 1u);           // round-to-nearest-even
  unsigned short h = (unsigned short)(u >> 16);
  return __builtin_bit_cast(__bf16, h);
}

// ---------------------------------------------------------------------------
// 0) One-time weight conversion: W[K,N] f32 (row major) -> WT[N,K] bf16.
//    Transposed so GEMM B-fragments are contiguous along K.
// ---------------------------------------------------------------------------
__global__ void convert_wT_kernel(const float* __restrict__ W, __bf16* __restrict__ WT,
                                  int K, int N) {
  size_t idx = (size_t)blockIdx.x * 256 + threadIdx.x;
  if (idx >= (size_t)K * N) return;
  int n = (int)(idx / K);
  int k = (int)(idx - (size_t)n * K);
  WT[idx] = f2bf(W[(size_t)k * N + n]);
}

// ---------------------------------------------------------------------------
// 1) Subword->token pooling: prefix matrix in LDS, row-normalize, pooled
//    attention mask + pooled embeddings (f32 + bf16 shadow).
// ---------------------------------------------------------------------------
__global__ void pool_kernel(const int* __restrict__ sub,    // [N,S,2]
                            const int* __restrict__ enc,    // [N,S]
                            const float* __restrict__ msk,  // [N,1]
                            const float* __restrict__ attn, // [N,S]
                            const float* __restrict__ emb,  // [V,D]
                            float* __restrict__ out_emb,    // [N,S,D]
                            __bf16* __restrict__ out_emb_bf,
                            float* __restrict__ out_attn) { // [N,S]
  int n = blockIdx.x;
  __shared__ float pfx[SS * SS];
  __shared__ float inv[SS];
  __shared__ int   senc[SS];
  int tid = threadIdx.x;
  for (int i = tid; i < SS * SS; i += 256) pfx[i] = 0.f;
  if (tid < SS) senc[tid] = enc[n * SS + tid];
  __syncthreads();
  float m = msk[n];
  if (tid < SS) {
    int r = sub[(n * SS + tid) * 2 + 0];
    int c = sub[(n * SS + tid) * 2 + 1];
    pfx[r * SS + c] = m;                      // .set semantics (idempotent)
  }
  __syncthreads();
  if (tid < SS) {
    float s = 0.f;
    for (int t = 0; t < SS; ++t) s += pfx[tid * SS + t];
    inv[tid] = 1.f / fmaxf(s, 1e-12f);
  }
  __syncthreads();
  for (int i = tid; i < SS * SS; i += 256) pfx[i] *= inv[i >> 6];
  __syncthreads();
  if (tid < SS) {
    float a = 0.f;
    for (int t = 0; t < SS; ++t) a += pfx[tid * SS + t] * attn[n * SS + t];
    out_attn[n * SS + tid] = a;
  }
  for (int i = tid; i < SS * DD; i += 256) {
    int s = i / DD, d = i - s * DD;
    float acc = 0.f;
    for (int t = 0; t < SS; ++t) {
      float w = pfx[s * SS + t];
      if (w != 0.f) acc += w * emb[(size_t)senc[t] * DD + d];
    }
    out_emb[(size_t)n * SS * DD + i] = acc;
    out_emb_bf[(size_t)n * SS * DD + i] = f2bf(acc);
  }
}

// ---------------------------------------------------------------------------
// 2) WMMA GEMM:  C[M,N] = act( A[M,K] @ W[K,N] + bias )
//    A: bf16 row-major [M,K].  WT: bf16 [N,K] (W transposed).
//    One wave computes a 64x16 slab: 4 row-tiles share one B fragment per
//    32-K step.  All fragment loads are contiguous bf16 vector loads; no
//    conversions in the K loop.  Out-of-range A rows are clamped (results
//    discarded by the guarded store).  Cf / Cbf are optional outputs.
//    act: 0=none, 1=tanh-gelu, 2=tanh.
// ---------------------------------------------------------------------------
__global__ void gemm_wmma_kernel(const __bf16* __restrict__ A,
                                 const __bf16* __restrict__ WT,
                                 const float* __restrict__ bias,
                                 float* __restrict__ Cf,
                                 __bf16* __restrict__ Cbf,
                                 int M, int N, int Kd, int act) {
  const int lane = threadIdx.x;        // 0..31
  const int half = lane >> 4;          // which 16-lane half
  const int l15  = lane & 15;
  const int col  = blockIdx.x * 16 + l15;   // B/C/D: N striped on lanes
  const int rowB = blockIdx.y * 64;

  const __bf16* Ap[4];
#pragma unroll
  for (int t = 0; t < 4; ++t) {
    int r = rowB + 16 * t + l15;            // A: row M striped on lanes
    Ap[t] = A + (size_t)(r < M ? r : (M - 1)) * Kd;   // clamp: no predication
  }
  const __bf16* Wrow = WT + (size_t)col * Kd;

  const v8f vzero = {0.f, 0.f, 0.f, 0.f, 0.f, 0.f, 0.f, 0.f};
  v8f acc[4];
#pragma unroll
  for (int t = 0; t < 4; ++t) acc[t] = vzero;

  for (int kb = 0; kb < Kd; kb += 32) {
    // B 32x16 bf16: lane holds column `col`; element e <-> K = kb+16*half+e
    // -> 16 contiguous bf16 (32B aligned) in the transposed weight row.
    v16bf bfv = *(const v16bf*)(Wrow + kb + 16 * half);
#pragma unroll
    for (int t = 0; t < 4; ++t) {
      // A 16x32 bf16 per lane: elements 0..7  <-> K kb+16*half+{0..7}
      //                        elements 8..15 <-> K kb+16*half+8+{0..7}
      v8bf alo = *(const v8bf*)(Ap[t] + kb + 16 * half);
      v8bf ahi = *(const v8bf*)(Ap[t] + kb + 16 * half + 8);
      v16bf af = __builtin_shufflevector(alo, ahi, 0, 1, 2, 3, 4, 5, 6, 7,
                                         8, 9, 10, 11, 12, 13, 14, 15);
      acc[t] = __builtin_amdgcn_wmma_f32_16x16x32_bf16(false, af, false, bfv,
                                                       (short)0, acc[t], false, false);
    }
  }

  float bs = bias ? bias[col] : 0.f;
#pragma unroll
  for (int t = 0; t < 4; ++t) {
#pragma unroll
    for (int r = 0; r < 8; ++r) {
      int m = rowB + 16 * t + r + 8 * half;   // C/D: VGPR r -> M=r / r+8 per half
      if (m < M) {
        float x = acc[t][r] + bs;
        if (act == 1) {                       // tanh-approx GELU (jax.nn.gelu default)
          float x3 = x * x * x;
          x = 0.5f * x * (1.f + tanhf(0.79788456080286535588f * (x + 0.044715f * x3)));
        } else if (act == 2) {
          x = tanhf(x);
        }
        if (Cf)  Cf[(size_t)m * N + col] = x;
        if (Cbf) Cbf[(size_t)m * N + col] = f2bf(x);
      }
    }
  }
}

// ---------------------------------------------------------------------------
// 3) Attention: one thread per (query row, head), flash-style online softmax.
//    Output written directly in bf16 (it only feeds the O-projection GEMM).
// ---------------------------------------------------------------------------
__global__ void attn_kernel(const float* __restrict__ Q, const float* __restrict__ Kb,
                            const float* __restrict__ V, const float* __restrict__ mask,
                            __bf16* __restrict__ O, int Nseq, int S) {
  int gr = blockIdx.x * blockDim.x + threadIdx.x;
  if (gr >= Nseq * S) return;
  int h = blockIdx.y;
  int n = gr / S;
  const float* qp = Q + (size_t)gr * DD + h * DH;
  float q[DH], o[DH];
#pragma unroll
  for (int d = 0; d < DH; ++d) { q[d] = qp[d]; o[d] = 0.f; }
  float mrun = -3.0e38f, l = 0.f;
  for (int j = 0; j < S; ++j) {
    const float* kp = Kb + ((size_t)n * S + j) * DD + h * DH;
    float dot = 0.f;
#pragma unroll
    for (int d = 0; d < DH; ++d) dot += q[d] * kp[d];
    float lg = dot * 0.125f + (mask[(size_t)n * S + j] > 0.f ? 0.f : -1e9f);
    float mn = fmaxf(mrun, lg);
    float corr = __expf(mrun - mn);
    float p = __expf(lg - mn);
    l = l * corr + p;
    const float* vp = V + ((size_t)n * S + j) * DD + h * DH;
#pragma unroll
    for (int d = 0; d < DH; ++d) o[d] = o[d] * corr + p * vp[d];
    mrun = mn;
  }
  float invl = 1.f / l;
  __bf16* op = O + (size_t)gr * DD + h * DH;
#pragma unroll
  for (int d = 0; d < DH; ++d) op[d] = f2bf(o[d] * invl);
}

// ---------------------------------------------------------------------------
// 4) out = LayerNorm(a + b), no scale/bias, eps 1e-12.  One block per row.
//    Optional bf16 shadow output (GEMM input).
// ---------------------------------------------------------------------------
__device__ __forceinline__ float block_reduce(float v, float* red) {
  int tid = threadIdx.x;
  red[tid] = v; __syncthreads();
  for (int s = 128; s > 0; s >>= 1) {
    if (tid < s) red[tid] += red[tid + s];
    __syncthreads();
  }
  float r = red[0]; __syncthreads();
  return r;
}

__global__ void add_ln_kernel(const float* __restrict__ a, const float* __restrict__ b,
                              float* __restrict__ out, __bf16* __restrict__ out_bf,
                              int rows) {
  __shared__ float red[256];
  int r = blockIdx.x;
  if (r >= rows) return;
  const float* pa = a + (size_t)r * DD;
  const float* pb = b + (size_t)r * DD;
  int tid = threadIdx.x;
  float v0 = pa[tid]       + pb[tid];
  float v1 = pa[tid + 256] + pb[tid + 256];
  float v2 = pa[tid + 512] + pb[tid + 512];
  float s  = block_reduce(v0 + v1 + v2, red);
  float mu = s * (1.f / 768.f);
  float d0 = v0 - mu, d1 = v1 - mu, d2 = v2 - mu;
  float s2 = block_reduce(d0 * d0 + d1 * d1 + d2 * d2, red);
  float inv = rsqrtf(s2 * (1.f / 768.f) + 1e-12f);
  float* po = out + (size_t)r * DD;
  po[tid] = d0 * inv; po[tid + 256] = d1 * inv; po[tid + 512] = d2 * inv;
  if (out_bf) {
    __bf16* pf = out_bf + (size_t)r * DD;
    pf[tid] = f2bf(d0 * inv); pf[tid + 256] = f2bf(d1 * inv); pf[tid + 512] = f2bf(d2 * inv);
  }
}

// ---------------------------------------------------------------------------
// Small glue kernels
// ---------------------------------------------------------------------------
__global__ void gather_rows_kernel(const float* __restrict__ src,
                                   float* __restrict__ dst, __bf16* __restrict__ dst_bf,
                                   int nrows, int stride_rows) {
  int i = blockIdx.x;
  if (i >= nrows) return;
  for (int d = threadIdx.x; d < DD; d += blockDim.x) {
    float v = src[(size_t)i * stride_rows * DD + d];
    dst[(size_t)i * DD + d] = v;
    dst_bf[(size_t)i * DD + d] = f2bf(v);
  }
}

__global__ void user_pool_kernel(const float* __restrict__ cls,   // [B*HIS, D]
                                 const float* __restrict__ qU,
                                 float* __restrict__ user0) {     // [B, D]
  int b = blockIdx.x;
  __shared__ float sc[HIS];
  __shared__ float w[HIS];
  int tid = threadIdx.x;
  if (tid < HIS) {
    const float* p = cls + (size_t)(b * HIS + tid) * DD;
    float s = 0.f;
    for (int d = 0; d < DD; ++d) s += p[d] * qU[d];
    sc[tid] = s * RT_D_INV;
  }
  __syncthreads();
  if (tid == 0) {
    float mx = -3e38f;
    for (int n = 0; n < HIS; ++n) mx = fmaxf(mx, sc[n]);
    float l = 0.f;
    for (int n = 0; n < HIS; ++n) { w[n] = __expf(sc[n] - mx); l += w[n]; }
    float invl = 1.f / l;
    for (int n = 0; n < HIS; ++n) w[n] *= invl;
  }
  __syncthreads();
  for (int d = tid; d < DD; d += blockDim.x) {
    float a = 0.f;
    for (int n = 0; n < HIS; ++n) a += w[n] * cls[(size_t)(b * HIS + n) * DD + d];
    user0[(size_t)b * DD + d] = a;
  }
}

__global__ void topk_kernel(const float* __restrict__ H,      // [B*HIS, S, D]
                            const float* __restrict__ user0,  // [B, D]
                            const float* __restrict__ attn,   // [B*HIS, S]
                            int* __restrict__ kid,            // [B*HIS, K]
                            float* __restrict__ psmask) {     // [B*HIS, K]
  int bn = blockIdx.x;
  int b  = bn / HIS;
  __shared__ float sc[SS];
  int s = threadIdx.x;                    // 64 threads
  const float* row = H + ((size_t)bn * SS + s) * DD;
  const float* u = user0 + (size_t)b * DD;
  float dot = 0.f;
  for (int d = 0; d < DD; ++d) dot += row[d] * u[d];
  float v = dot * RT_D_INV;
  if (attn[bn * SS + s] <= 0.f) v = -1e9f;
  sc[s] = v;
  __syncthreads();
  if (s == 0) {                           // jax.lax.top_k: descending, ties -> lower idx
    for (int j = 0; j < TOPK; ++j) {
      float best = -3.4e38f; int bi = 0;
      for (int t = 0; t < SS; ++t)
        if (sc[t] > best) { best = sc[t]; bi = t; }
      kid[bn * TOPK + j] = bi;
      psmask[bn * TOPK + j] = attn[bn * SS + bi];
      sc[bi] = -3.4e38f;
    }
  }
}

__global__ void gather_terms_kernel(const float* __restrict__ H,  // [B*HIS, S, D]
                                    const int* __restrict__ kid,  // [B*HIS, K]
                                    float* __restrict__ out,      // [B, US, D]
                                    __bf16* __restrict__ out_bf) {
  size_t idx = (size_t)blockIdx.x * 256 + threadIdx.x;
  const size_t total = (size_t)BB * US * DD;
  if (idx >= total) return;
  int d  = (int)(idx % DD);
  int r  = (int)(idx / DD);               // b*US + n*K + j
  int b  = r / US;
  int nj = r - b * US;
  int n  = nj / TOPK;
  int j  = nj - n * TOPK;
  int s  = kid[(b * HIS + n) * TOPK + j];
  float v = H[(((size_t)(b * HIS + n)) * SS + s) * DD + d];
  out[idx] = v;
  out_bf[idx] = f2bf(v);
}

__global__ void final_kernel(const float* __restrict__ cddR,   // [B*CDD, D]
                             const float* __restrict__ userR,  // [B, D]
                             float* __restrict__ prob) {       // [B, CDD]
  int b = blockIdx.x;
  __shared__ float sc[CDD];
  int tid = threadIdx.x;
  if (tid < CDD) {
    const float* c = cddR + (size_t)(b * CDD + tid) * DD;
    const float* u = userR + (size_t)b * DD;
    float s = 0.f;
    for (int d = 0; d < DD; ++d) s += c[d] * u[d];
    sc[tid] = s * RT_D_INV;
  }
  __syncthreads();
  if (tid == 0) {
    float mx = -3e38f;
    for (int c = 0; c < CDD; ++c) mx = fmaxf(mx, sc[c]);
    float l = 0.f;
    for (int c = 0; c < CDD; ++c) l += __expf(sc[c] - mx);
    float lse = __logf(l);
    for (int c = 0; c < CDD; ++c) prob[b * CDD + c] = sc[c] - mx - lse;
  }
}

// ---------------------------------------------------------------------------
// Host-side orchestration
// ---------------------------------------------------------------------------
static inline void launch_gemm(const __bf16* A, const __bf16* WT, const float* bias,
                               float* Cf, __bf16* Cbf,
                               int M, int N, int Kd, int act, hipStream_t s) {
  dim3 g(N / 16, (M + 63) / 64);
  gemm_wmma_kernel<<<g, 32, 0, s>>>(A, WT, bias, Cf, Cbf, M, N, Kd, act);
}

static inline void launch_convert(const float* W, __bf16* WT, int K, int N,
                                  hipStream_t s) {
  size_t total = (size_t)K * N;
  convert_wT_kernel<<<(unsigned)((total + 255) / 256), 256, 0, s>>>(W, WT, K, N);
}

static void run_transformer(const __bf16* WqT, const __bf16* WkT, const __bf16* WvT,
                            const __bf16* WoT, const __bf16* W1T, const __bf16* W2T,
                            const float* x, const __bf16* xbf, const float* mask,
                            float* q, float* k, float* v,
                            __bf16* hbf, __bf16* aobf, __bf16* h1bf,
                            float* out, int Nseq, int S, hipStream_t stream) {
  int M = Nseq * S;
  launch_gemm(xbf, WqT, nullptr, q, nullptr, M, DD, DD, 0, stream);
  launch_gemm(xbf, WkT, nullptr, k, nullptr, M, DD, DD, 0, stream);
  launch_gemm(xbf, WvT, nullptr, v, nullptr, M, DD, DD, 0, stream);
  dim3 ag((M + 255) / 256, NHH);
  attn_kernel<<<ag, 256, 0, stream>>>(q, k, v, mask, aobf, Nseq, S);
  launch_gemm(aobf, WoT, nullptr, q, nullptr, M, DD, DD, 0, stream);  // o-proj -> q
  add_ln_kernel<<<M, 256, 0, stream>>>(x, q, k, hbf, M);              // h = LN(x+o)
  launch_gemm(hbf, W1T, nullptr, nullptr, h1bf, M, FFF, DD, 1, stream); // gelu -> bf16
  launch_gemm(h1bf, W2T, nullptr, v, nullptr, M, DD, FFF, 0, stream);
  add_ln_kernel<<<M, 256, 0, stream>>>(k, v, out, nullptr, M);        // LN(h + ffn)
}

extern "C" void kernel_launch(void* const* d_in, const int* in_sizes, int n_in,
                              void* d_out, int out_size, void* d_ws, size_t ws_size,
                              hipStream_t stream) {
  // ---- inputs (setup_inputs dict order) ----
  const int*   cdd_sub   = (const int*)d_in[0];
  const int*   his_sub   = (const int*)d_in[1];
  const int*   cdd_enc   = (const int*)d_in[2];
  const int*   his_enc   = (const int*)d_in[3];
  const float* cdd_msk   = (const float*)d_in[4];
  const float* his_msk   = (const float*)d_in[5];
  const float* cdd_amask = (const float*)d_in[6];
  const float* his_amask = (const float*)d_in[7];
  const float* emb       = (const float*)d_in[8];
  const float* bWq = (const float*)d_in[9];
  const float* bWk = (const float*)d_in[10];
  const float* bWv = (const float*)d_in[11];
  const float* bWo = (const float*)d_in[12];
  const float* bW1 = (const float*)d_in[13];
  const float* bW2 = (const float*)d_in[14];
  const float* eWq = (const float*)d_in[15];
  const float* eWk = (const float*)d_in[16];
  const float* eWv = (const float*)d_in[17];
  const float* eWo = (const float*)d_in[18];
  const float* eW1 = (const float*)d_in[19];
  const float* eW2 = (const float*)d_in[20];
  const float* qU  = (const float*)d_in[21];
  const float* pW  = (const float*)d_in[22];
  const float* pb  = (const float*)d_in[23];

  // ---- workspace layout ----
  const size_t R = (size_t)BB * HIS * SS;          // 25600 max rows
  float* ws = (float*)d_ws;
  // f32 region
  float* bx    = ws;                  // R*D
  float* bq    = bx    + R * DD;      // R*D
  float* bk    = bq    + R * DD;      // R*D
  float* bv    = bk    + R * DD;      // R*D
  float* bhid  = bv    + R * DD;      // R*D (his hidden, persists)
  float* aC    = bhid  + R * DD;      // [40*64]
  float* aH    = aC    + BB * CDD * SS;            // [400*64]
  float* bclsF = aH    + BB * HIS * SS;            // [400, D]
  float* cddR  = bclsF + (size_t)BB * HIS * DD;    // [40, D]
  float* usrR  = cddR  + (size_t)BB * CDD * DD;    // [8, D]
  float* usr0  = usrR  + (size_t)BB * DD;          // [8, D]
  float* psm   = usr0  + (size_t)BB * DD;          // [3200]
  // bf16 region (all offsets multiples of 16 elements -> 32B aligned)
  __bf16* bf   = (__bf16*)(psm + BB * HIS * TOPK);
  __bf16* xbf   = bf;                        // R*D
  __bf16* hbf   = xbf   + R * DD;            // R*D
  __bf16* aobf  = hbf   + R * DD;            // R*D
  __bf16* h1bf  = aobf  + R * DD;            // R*FF
  __bf16* clsbf = h1bf  + R * FFF;           // [400, D]
  __bf16* wts   = clsbf + (size_t)BB * HIS * DD;
  const size_t WDD = (size_t)DD * DD, WDF = (size_t)DD * FFF;
  __bf16* bWqT = wts;            __bf16* bWkT = bWqT + WDD;
  __bf16* bWvT = bWkT + WDD;     __bf16* bWoT = bWvT + WDD;
  __bf16* bW1T = bWoT + WDD;     __bf16* bW2T = bW1T + WDF;
  __bf16* eWqT = bW2T + WDF;     __bf16* eWkT = eWqT + WDD;
  __bf16* eWvT = eWkT + WDD;     __bf16* eWoT = eWvT + WDD;
  __bf16* eW1T = eWoT + WDD;     __bf16* eW2T = eW1T + WDF;
  __bf16* pWT  = eW2T + WDF;

  float* prob    = (float*)d_out;              // [B, CDD] = 40 floats
  int*   kid_out = ((int*)d_out) + BB * CDD;   // [B, HIS, K] = 3200 ints

  // ---- 0) convert all weights to transposed bf16 ----
  launch_convert(bWq, bWqT, DD, DD, stream);
  launch_convert(bWk, bWkT, DD, DD, stream);
  launch_convert(bWv, bWvT, DD, DD, stream);
  launch_convert(bWo, bWoT, DD, DD, stream);
  launch_convert(bW1, bW1T, DD, FFF, stream);
  launch_convert(bW2, bW2T, FFF, DD, stream);
  launch_convert(eWq, eWqT, DD, DD, stream);
  launch_convert(eWk, eWkT, DD, DD, stream);
  launch_convert(eWv, eWvT, DD, DD, stream);
  launch_convert(eWo, eWoT, DD, DD, stream);
  launch_convert(eW1, eW1T, DD, FFF, stream);
  launch_convert(eW2, eW2T, FFF, DD, stream);
  launch_convert(pW,  pWT,  DD, DD, stream);

  // ---- 1) candidate pipeline ----
  pool_kernel<<<BB * CDD, 256, 0, stream>>>(cdd_sub, cdd_enc, cdd_msk, cdd_amask,
                                            emb, bx, xbf, aC);
  run_transformer(bWqT, bWkT, bWvT, bWoT, bW1T, bW2T, bx, xbf, aC,
                  bq, bk, bv, hbf, aobf, h1bf, bx, BB * CDD, SS, stream);
  gather_rows_kernel<<<BB * CDD, 256, 0, stream>>>(bx, bclsF, clsbf, BB * CDD, SS);
  launch_gemm(clsbf, pWT, pb, cddR, nullptr, BB * CDD, DD, DD, 2, stream);  // tanh proj

  // ---- 2) history pipeline ----
  pool_kernel<<<BB * HIS, 256, 0, stream>>>(his_sub, his_enc, his_msk, his_amask,
                                            emb, bx, xbf, aH);
  run_transformer(eWqT, eWkT, eWvT, eWoT, eW1T, eW2T, bx, xbf, aH,
                  bq, bk, bv, hbf, aobf, h1bf, bhid, BB * HIS, SS, stream);
  gather_rows_kernel<<<BB * HIS, 256, 0, stream>>>(bhid, bclsF, clsbf, BB * HIS, SS);

  // ---- 3) user attention pooling over history CLS ----
  user_pool_kernel<<<BB, 256, 0, stream>>>(bclsF, qU, usr0);

  // ---- 4) personalized term top-K selection ----
  topk_kernel<<<BB * HIS, SS, 0, stream>>>(bhid, usr0, aH, kid_out, psm);
  {
    size_t total = (size_t)BB * US * DD;
    gather_terms_kernel<<<(unsigned)((total + 255) / 256), 256, 0, stream>>>(
        bhid, kid_out, bx, xbf);
  }

  // ---- 5) user BERT over selected terms ----
  run_transformer(bWqT, bWkT, bWvT, bWoT, bW1T, bW2T, bx, xbf, psm,
                  bq, bk, bv, hbf, aobf, h1bf, bx, BB, US, stream);
  gather_rows_kernel<<<BB, 256, 0, stream>>>(bx, bclsF, clsbf, BB, US);
  launch_gemm(clsbf, pWT, pb, usrR, nullptr, BB, DD, DD, 2, stream);        // tanh proj

  // ---- 6) final scoring + log-softmax ----
  final_kernel<<<BB, 32, 0, stream>>>(cddR, usrR, prob);
}
